// PRNN_38208029065628
// MI455X (gfx1250) — matrix operand order repaired and checked by reading, not compiled, not run on hardware
//
#include <hip/hip_runtime.h>

typedef _Float16 h16;
typedef __attribute__((ext_vector_type(16))) _Float16 v16h;
typedef __attribute__((ext_vector_type(8)))  _Float16 v8h;
typedef __attribute__((ext_vector_type(8)))  float    v8f;
typedef __attribute__((ext_vector_type(4)))  float    v4f;

#define T_STEPS 1024
#define BATCH   256
#define IDIM    64
#define HDIM    512

#define IMG_BYTES 784            // one 16x16 f16 act tile: 16 rows * 48B (32B payload + 16B pad) + 16B tile pad
#define BUF_BYTES (32 * IMG_BYTES)

// ---------- prep: fp32 -> f16 weight images, fused bias ----------
__global__ void prnn_prep(const float* __restrict__ Win,
                          const float* __restrict__ bin,
                          const float* __restrict__ Whh,
                          const float* __restrict__ bhh,
                          h16* __restrict__ WhhH,
                          h16* __restrict__ WinH,
                          float* __restrict__ bias)
{
    int i = blockIdx.x * 256 + threadIdx.x;
    if (i < HDIM * HDIM) WhhH[i] = (h16)Whh[i];
    if (i < HDIM * IDIM) WinH[i] = (h16)Win[i];
    if (i < HDIM)        bias[i] = bin[i] + bhh[i];
}

__device__ __forceinline__ float photonic(float h) {
    float a  = __builtin_fmaf(0.25f, h, 0.3f);
    float x2 = h * h;
    return x2 * __builtin_amdgcn_rcpf(__builtin_fmaf(a, a, x2));
}

__device__ __forceinline__ v16h cat8(v8h lo, v8h hi) {
    return __builtin_shufflevector(lo, hi, 0,1,2,3,4,5,6,7,8,9,10,11,12,13,14,15);
}

__device__ __forceinline__ v16h loadA(const h16* p) {
    return cat8(*(const v8h*)p, *(const v8h*)(p + 8));
}

// Zero the compiler cannot prove constant across iterations: blocks LICM of
// derived loads while preserving pointer provenance (global_load, not flat).
__device__ __forceinline__ int opq0() {
    int v = 0;
    asm volatile("" : "+v"(v));
    return v;
}

// One WG (512 threads = 16 waves) per 16-batch tile. Wave w owns hidden
// columns j in [32w, 32w+32): two j-tiles -> two independent WMMA chains
// sharing every B operand. h kept in registers (D layout); act(h) exchanged
// via double-buffered LDS as ready-made f16 B-operand tiles.
__launch_bounds__(512, 1)
__global__ void prnn_main(const float* __restrict__ x,
                          const float* __restrict__ h0,
                          const h16*  __restrict__ WhhH,
                          const h16*  __restrict__ WinH,
                          const float* __restrict__ bias,
                          float* __restrict__ out)
{
    __shared__ __align__(16) unsigned char lds[2 * BUF_BYTES];

    const int lane = threadIdx.x & 31;
    const int wv   = threadIdx.x >> 5;     // 0..15
    const int lm   = lane & 15;
    const int hi   = lane >> 4;            // lane half: 0/1
    const int j0   = wv << 5;              // 32 columns per wave
    const int b0   = blockIdx.x << 4;

    // A-operand row pointers (A: lane row M = lane%16) for the two j-tiles.
    const h16* whhRow0 = WhhH + (j0 + lm) * HDIM + hi * 16;
    const h16* whhRow1 = whhRow0 + 16 * HDIM;
    const h16* winRow0 = WinH + (j0 + lm) * IDIM + hi * 16;
    const h16* winRow1 = winRow0 + 16 * IDIM;

    // Preload W_in A-operand tiles (K = 0..31, 32..63) x two j-tiles.
    v16h Ain[2][2];
    #pragma unroll
    for (int kt = 0; kt < 2; ++kt) {
        Ain[kt][0] = loadA(winRow0 + kt * 32);
        Ain[kt][1] = loadA(winRow1 + kt * 32);
    }

    // Fused bias, D layout (reg r -> j = j0 + 16*c + hi*8 + r).
    float bv[2][8];
    #pragma unroll
    for (int r = 0; r < 8; ++r) {
        bv[0][r] = bias[j0 + hi * 8 + r];
        bv[1][r] = bias[j0 + 16 + hi * 8 + r];
    }

    // h state, D layout: h[c][r] holds (j = j0 + 16c + hi*8 + r, b = b0 + lm).
    float h[2][8];
    {
        const float* hp = h0 + (b0 + lm) * HDIM + j0 + hi * 8;
        #pragma unroll
        for (int r = 0; r < 8; ++r) { h[0][r] = hp[r]; h[1][r] = hp[16 + r]; }
    }

    const int myStore = (2 * wv) * IMG_BYTES + lm * 48 + hi * 16;
    const int myBLoad = hi * IMG_BYTES + lm * 48;

    #pragma unroll 1
    for (int t = 0; t < T_STEPS; ++t) {
        // ---- top of step: issue ALL global loads early (no LDS deps) so
        // their latency hides behind act/output/publish and the barrier. ----
        const h16* wr0 = whhRow0 + opq0();
        const h16* wr1 = whhRow1 + opq0();

        v16h A0b[4], A1b[4];
        #pragma unroll
        for (int p = 0; p < 3; ++p) {
            A0b[p] = loadA(wr0 + p * 32);
            A1b[p] = loadA(wr1 + p * 32);
        }

        const float* xp = x + (size_t)t * (BATCH * IDIM) + (b0 + lm) * IDIM + hi * 16;
        if (t + 1 < T_STEPS) __builtin_prefetch(xp + BATCH * IDIM, 0, 1);
        v4f xv[2][4];
        #pragma unroll
        for (int kt = 0; kt < 2; ++kt) {
            const v4f* xq = (const v4f*)(xp + kt * 32);
            #pragma unroll
            for (int q = 0; q < 4; ++q) xv[kt][q] = xq[q];
        }

        // act(h_t)
        float actf[2][8];
        #pragma unroll
        for (int r = 0; r < 8; ++r) {
            actf[0][r] = photonic(h[0][r]);
            actf[1][r] = photonic(h[1][r]);
        }

        // output[t-1] = act(h_t): write-once stream -> nontemporal stores.
        if (t > 0) {
            float* o = out + (size_t)(t - 1) * (BATCH * HDIM)
                           + (b0 + lm) * HDIM + j0 + hi * 8;
            __builtin_nontemporal_store(
                (v4f){ actf[0][0], actf[0][1], actf[0][2], actf[0][3] }, (v4f*)o);
            __builtin_nontemporal_store(
                (v4f){ actf[0][4], actf[0][5], actf[0][6], actf[0][7] }, (v4f*)(o + 4));
            __builtin_nontemporal_store(
                (v4f){ actf[1][0], actf[1][1], actf[1][2], actf[1][3] }, (v4f*)(o + 16));
            __builtin_nontemporal_store(
                (v4f){ actf[1][4], actf[1][5], actf[1][6], actf[1][7] }, (v4f*)(o + 20));
        }

        // Publish act(h_t) as f16 B-operand images (D layout == B layout slice).
        {
            unsigned char* st = lds + (t & 1) * BUF_BYTES + myStore;
            v8h pk0, pk1;
            #pragma unroll
            for (int r = 0; r < 8; ++r) {
                pk0[r] = (h16)actf[0][r];
                pk1[r] = (h16)actf[1][r];
            }
            *(v8h*)st               = pk0;
            *(v8h*)(st + IMG_BYTES) = pk1;
        }

        // Convert x to f16 B operands while waiting.
        v16h Bx[2];
        #pragma unroll
        for (int kt = 0; kt < 2; ++kt) {
            #pragma unroll
            for (int c = 0; c < 4; ++c) {
                Bx[kt][c]      = (h16)xv[kt][0][c];
                Bx[kt][4 + c]  = (h16)xv[kt][1][c];
                Bx[kt][8 + c]  = (h16)xv[kt][2][c];
                Bx[kt][12 + c] = (h16)xv[kt][3][c];
            }
        }

        __syncthreads();   // double buffer -> single barrier per step

        // pre^T accumulation: acc = bias + W_in x_t^T + W_hh act^T
        v8f acc0, acc1;
        #pragma unroll
        for (int r = 0; r < 8; ++r) { acc0[r] = bv[0][r]; acc1[r] = bv[1][r]; }

        const unsigned char* bpB = lds + (t & 1) * BUF_BYTES + myBLoad;
        v16h Bhb[2];
        Bhb[0] = cat8(*(const v8h*)bpB, *(const v8h*)(bpB + 16));

        // Input part overlaps the first B-tile's DS latency.
        #pragma unroll
        for (int kt = 0; kt < 2; ++kt) {
            acc0 = __builtin_amdgcn_wmma_f32_16x16x32_f16(false, Ain[kt][0], false, Bx[kt],
                                                          (short)0, acc0, false, false);
            acc1 = __builtin_amdgcn_wmma_f32_16x16x32_f16(false, Ain[kt][1], false, Bx[kt],
                                                          (short)0, acc1, false, false);
        }

        // Recurrent part: K = 512 in 16 tiles of 32.
        // A ring is 4 deep (global, ~3 tiles in flight), B ring 2 deep (LDS).
        // sched_barrier(0) pins prefetch issue ahead of each WMMA pair so the
        // waits become partial instead of full drains.
        #pragma unroll
        for (int kt = 0; kt < 16; ++kt) {
            if (kt + 3 < 16) {
                A0b[(kt + 3) & 3] = loadA(wr0 + (kt + 3) * 32);
                A1b[(kt + 3) & 3] = loadA(wr1 + (kt + 3) * 32);
            }
            if (kt + 1 < 16) {
                const unsigned char* bp = bpB + (size_t)(kt + 1) * (2 * IMG_BYTES);
                Bhb[(kt + 1) & 1] = cat8(*(const v8h*)bp, *(const v8h*)(bp + 16));
            }
            __builtin_amdgcn_sched_barrier(0);
            acc0 = __builtin_amdgcn_wmma_f32_16x16x32_f16(false, A0b[kt & 3], false, Bhb[kt & 1],
                                                          (short)0, acc0, false, false);
            acc1 = __builtin_amdgcn_wmma_f32_16x16x32_f16(false, A1b[kt & 3], false, Bhb[kt & 1],
                                                          (short)0, acc1, false, false);
        }

        // h_{t+1} = 0.5 h + 0.5 pre
        #pragma unroll
        for (int r = 0; r < 8; ++r) {
            h[0][r] = 0.5f * h[0][r] + 0.5f * acc0[r];
            h[1][r] = 0.5f * h[1][r] + 0.5f * acc1[r];
        }
    }

    // Tail: output[T-1] = act(h_T); h_final = h_T.
    {
        float actf[2][8];
        #pragma unroll
        for (int r = 0; r < 8; ++r) {
            actf[0][r] = photonic(h[0][r]);
            actf[1][r] = photonic(h[1][r]);
        }
        float* o = out + (size_t)(T_STEPS - 1) * (BATCH * HDIM)
                       + (b0 + lm) * HDIM + j0 + hi * 8;
        *(v4f*)o        = (v4f){ actf[0][0], actf[0][1], actf[0][2], actf[0][3] };
        *(v4f*)(o + 4)  = (v4f){ actf[0][4], actf[0][5], actf[0][6], actf[0][7] };
        *(v4f*)(o + 16) = (v4f){ actf[1][0], actf[1][1], actf[1][2], actf[1][3] };
        *(v4f*)(o + 20) = (v4f){ actf[1][4], actf[1][5], actf[1][6], actf[1][7] };

        float* hf = out + (size_t)T_STEPS * (BATCH * HDIM)
                        + (b0 + lm) * HDIM + j0 + hi * 8;
        *(v4f*)hf        = (v4f){ h[0][0], h[0][1], h[0][2], h[0][3] };
        *(v4f*)(hf + 4)  = (v4f){ h[0][4], h[0][5], h[0][6], h[0][7] };
        *(v4f*)(hf + 16) = (v4f){ h[1][0], h[1][1], h[1][2], h[1][3] };
        *(v4f*)(hf + 20) = (v4f){ h[1][4], h[1][5], h[1][6], h[1][7] };
    }
}

extern "C" void kernel_launch(void* const* d_in, const int* in_sizes, int n_in,
                              void* d_out, int out_size, void* d_ws, size_t ws_size,
                              hipStream_t stream)
{
    const float* x    = (const float*)d_in[0];
    const float* h0   = (const float*)d_in[1];
    const float* Win  = (const float*)d_in[2];
    const float* bin  = (const float*)d_in[3];
    const float* Whh  = (const float*)d_in[4];
    const float* bhh  = (const float*)d_in[5];
    float* out = (float*)d_out;

    unsigned char* ws = (unsigned char*)d_ws;
    h16*   WhhH = (h16*)(ws);                      // 512*512*2 = 524288 B
    h16*   WinH = (h16*)(ws + 524288);             // 512*64*2  =  65536 B
    float* bias = (float*)(ws + 524288 + 65536);   // 512*4     =   2048 B

    prnn_prep<<<(HDIM * HDIM + 255) / 256, 256, 0, stream>>>(Win, bin, Whh, bhh,
                                                             WhhH, WinH, bias);
    prnn_main<<<BATCH / 16, 512, 0, stream>>>(x, h0, WhhH, WinH, bias, out);
}
